// PointNet_79843442032743
// MI455X (gfx1250) — matrix operand order repaired and checked by reading, not compiled, not run on hardware
//
#include <hip/hip_runtime.h>

typedef __attribute__((ext_vector_type(16))) _Float16 v16h;
typedef __attribute__((ext_vector_type(8)))  _Float16 v8h;
typedef __attribute__((ext_vector_type(8)))  float    v8f;

static __device__ __forceinline__ v8f wmma16(v16h a, v16h b, v8f c) {
  return __builtin_amdgcn_wmma_f32_16x16x32_f16(false, a, false, b, (short)0, c,
                                                false, false);
}

static __device__ __forceinline__ v8f zero8() {
  v8f z;
#pragma unroll
  for (int i = 0; i < 8; ++i) z[i] = 0.0f;
  return z;
}

static __device__ __forceinline__ v16h zero16h() {
  v16h z;
#pragma unroll
  for (int i = 0; i < 16; ++i) z[i] = (_Float16)0.0f;
  return z;
}

__global__ void zero_kernel(float* __restrict__ p, long n) {
  long i = (long)blockIdx.x * blockDim.x + threadIdx.x;
  long stride = (long)gridDim.x * blockDim.x;
  for (; i < n; i += stride) p[i] = 0.0f;
}

// ---------------------------------------------------------------------------
// Layer 1: x = [pos_j(2), pos_j - pos_i(2)] (K padded 4->32)
//   m = relu(x@w1a+b1a)@w1b+b1b ; h1[dst] = max(h1[dst], m)  (h1 init 0)
// One wave handles a tile of 16 edges; N=32 split across two WMMA accumulators.
// ---------------------------------------------------------------------------
__global__ __launch_bounds__(256) void layer1_kernel(
    const float* __restrict__ pos,
    const float* __restrict__ w1a, const float* __restrict__ b1a,
    const float* __restrict__ w1b, const float* __restrict__ b1b,
    const int* __restrict__ esrc, const int* __restrict__ edst,
    float* __restrict__ h1, int ntiles, int E)
{
  __shared__ __align__(16) _Float16 stage[8][16 * 32];
  __shared__ __align__(16) int dstbuf[8][16];

  const int lane   = threadIdx.x & 31;
  const int wv     = threadIdx.x >> 5;
  const int col    = lane & 15;
  const int half16 = lane >> 4;     // which half-wave
  const int kb     = half16 * 16;   // B-fragment K base (lanes 0-15: K 0-15)
  const int mbase  = half16 * 8;    // D-layout row base
  const int kA     = half16 * 8;    // A-layout K chunk base

  // --- B fragment for GEMM1: w1a (4x32), K zero-padded to 32 ---
  v16h Wa_lo = zero16h(), Wa_hi = zero16h();
  if (half16 == 0) {
#pragma unroll
    for (int k = 0; k < 4; ++k) {
      Wa_lo[k] = (_Float16)w1a[k * 32 + col];
      Wa_hi[k] = (_Float16)w1a[k * 32 + col + 16];
    }
  }
  // --- B fragments for GEMM2: w1b (32x32) ---
  v16h Wb_lo, Wb_hi;
#pragma unroll
  for (int k = 0; k < 16; ++k) {
    Wb_lo[k] = (_Float16)w1b[(kb + k) * 32 + col];
    Wb_hi[k] = (_Float16)w1b[(kb + k) * 32 + col + 16];
  }
  const float bias1_lo = b1a[col], bias1_hi = b1a[col + 16];
  const float bias2_lo = b1b[col], bias2_hi = b1b[col + 16];

  _Float16* st = stage[wv];
  int* db = dstbuf[wv];

  const int wave = blockIdx.x * 8 + wv;
  const int nwaves = gridDim.x * 8;
  for (int tile = wave; tile < ntiles; tile += nwaves) {
    int e = tile * 16 + col;
    if (e >= E) e = E - 1;            // duplicate edges are idempotent under max
    const int s = esrc[e];
    const int d = edst[e];
    const float sx = pos[2 * s], sy = pos[2 * s + 1];
    const float px = pos[2 * d], py = pos[2 * d + 1];
    if (half16 == 0) db[col] = d;

    // A fragment: row M=col, features K0..3 = [sx, sy, sx-px, sy-py]
    v16h A = zero16h();
    if (half16 == 0) {
      A[0] = (_Float16)sx;        A[1] = (_Float16)sy;
      A[2] = (_Float16)(sx - px); A[3] = (_Float16)(sy - py);
    }
    v8f t0 = wmma16(A, Wa_lo, zero8());
    v8f t1 = wmma16(A, Wa_hi, zero8());

    // bias + relu, stage as f16 in row-major [16][32]
#pragma unroll
    for (int r = 0; r < 8; ++r) {
      float u0 = t0[r] + bias1_lo; u0 = u0 > 0.0f ? u0 : 0.0f;
      float u1 = t1[r] + bias1_hi; u1 = u1 > 0.0f ? u1 : 0.0f;
      st[(mbase + r) * 32 + col]      = (_Float16)u0;
      st[(mbase + r) * 32 + col + 16] = (_Float16)u1;
    }
    __builtin_amdgcn_wave_barrier();
    asm volatile("s_wait_dscnt 0x0" ::: "memory");

    // reload per A layout: row M=col, K {kb..kb+15}
    v8h lo = *(const v8h*)(st + col * 32 + kb);
    v8h hi = *(const v8h*)(st + col * 32 + kb + 8);
    v16h A2;
#pragma unroll
    for (int i = 0; i < 8; ++i) { A2[i] = lo[i]; A2[8 + i] = hi[i]; }
    int4 di0 = *(const int4*)(db + mbase);
    int4 di1 = *(const int4*)(db + mbase + 4);
    __builtin_amdgcn_wave_barrier();

    v8f d0 = wmma16(A2, Wb_lo, zero8());
    v8f d1 = wmma16(A2, Wb_hi, zero8());

    int dr[8] = {di0.x, di0.y, di0.z, di0.w, di1.x, di1.y, di1.z, di1.w};
#pragma unroll
    for (int r = 0; r < 8; ++r) {
      float u0 = d0[r] + bias2_lo;
      float u1 = d1[r] + bias2_hi;
      unsigned* row = (unsigned*)(h1 + (long)dr[r] * 32);
      if (u0 > 0.0f) atomicMax(row + col,      __float_as_uint(u0));
      if (u1 > 0.0f) atomicMax(row + col + 16, __float_as_uint(u1));
    }
  }
}

// ---------------------------------------------------------------------------
// Layer 2: x = [h1_j(32), pos_j - pos_i(2)] (K 34 -> two K-blocks of 32)
// ---------------------------------------------------------------------------
__global__ __launch_bounds__(256) void layer2_kernel(
    const float* __restrict__ pos, const float* __restrict__ h1,
    const float* __restrict__ w2a, const float* __restrict__ b2a,
    const float* __restrict__ w2b, const float* __restrict__ b2b,
    const int* __restrict__ esrc, const int* __restrict__ edst,
    float* __restrict__ h2, int ntiles, int E)
{
  __shared__ __align__(16) _Float16 stage[8][16 * 32];
  __shared__ __align__(16) int dstbuf[8][16];

  const int lane   = threadIdx.x & 31;
  const int wv     = threadIdx.x >> 5;
  const int col    = lane & 15;
  const int half16 = lane >> 4;
  const int kb     = half16 * 16;
  const int mbase  = half16 * 8;
  const int kA     = half16 * 8;

  // B fragments: GEMM1 K-block0 = w2a rows 0..31
  v16h Wa0_lo, Wa0_hi;
#pragma unroll
  for (int k = 0; k < 16; ++k) {
    Wa0_lo[k] = (_Float16)w2a[(kb + k) * 32 + col];
    Wa0_hi[k] = (_Float16)w2a[(kb + k) * 32 + col + 16];
  }
  // GEMM1 K-block1 = w2a rows 32..33 (zero-padded to 32)
  v16h Wa1_lo = zero16h(), Wa1_hi = zero16h();
  if (half16 == 0) {
#pragma unroll
    for (int k = 0; k < 2; ++k) {
      Wa1_lo[k] = (_Float16)w2a[(32 + k) * 32 + col];
      Wa1_hi[k] = (_Float16)w2a[(32 + k) * 32 + col + 16];
    }
  }
  // GEMM2: w2b (32x32)
  v16h Wb_lo, Wb_hi;
#pragma unroll
  for (int k = 0; k < 16; ++k) {
    Wb_lo[k] = (_Float16)w2b[(kb + k) * 32 + col];
    Wb_hi[k] = (_Float16)w2b[(kb + k) * 32 + col + 16];
  }
  const float bias1_lo = b2a[col], bias1_hi = b2a[col + 16];
  const float bias2_lo = b2b[col], bias2_hi = b2b[col + 16];

  _Float16* st = stage[wv];
  int* db = dstbuf[wv];

  const int wave = blockIdx.x * 8 + wv;
  const int nwaves = gridDim.x * 8;
  for (int tile = wave; tile < ntiles; tile += nwaves) {
    int e = tile * 16 + col;
    if (e >= E) e = E - 1;
    const int s = esrc[e];
    const int d = edst[e];
    if (half16 == 0) db[col] = d;

    // A K-block0: gather h1[src] directly in A layout (row M=col)
    const float* hs = h1 + (long)s * 32;
    float4 f0 = *(const float4*)(hs + kb);
    float4 f1 = *(const float4*)(hs + kb + 4);
    float4 f2 = *(const float4*)(hs + kb + 8);
    float4 f3 = *(const float4*)(hs + kb + 12);
    v16h A0;
    A0[0]  = (_Float16)f0.x; A0[1]  = (_Float16)f0.y;
    A0[2]  = (_Float16)f0.z; A0[3]  = (_Float16)f0.w;
    A0[4]  = (_Float16)f1.x; A0[5]  = (_Float16)f1.y;
    A0[6]  = (_Float16)f1.z; A0[7]  = (_Float16)f1.w;
    A0[8]  = (_Float16)f2.x; A0[9]  = (_Float16)f2.y;
    A0[10] = (_Float16)f2.z; A0[11] = (_Float16)f2.w;
    A0[12] = (_Float16)f3.x; A0[13] = (_Float16)f3.y;
    A0[14] = (_Float16)f3.z; A0[15] = (_Float16)f3.w;

    // A K-block1: features 32,33 = pos_j - pos_i, rest zero
    v16h A1 = zero16h();
    if (half16 == 0) {
      float sx = pos[2 * s], sy = pos[2 * s + 1];
      float px = pos[2 * d], py = pos[2 * d + 1];
      A1[0] = (_Float16)(sx - px);
      A1[1] = (_Float16)(sy - py);
    }

    v8f t0 = wmma16(A0, Wa0_lo, zero8());
    t0     = wmma16(A1, Wa1_lo, t0);
    v8f t1 = wmma16(A0, Wa0_hi, zero8());
    t1     = wmma16(A1, Wa1_hi, t1);

#pragma unroll
    for (int r = 0; r < 8; ++r) {
      float u0 = t0[r] + bias1_lo; u0 = u0 > 0.0f ? u0 : 0.0f;
      float u1 = t1[r] + bias1_hi; u1 = u1 > 0.0f ? u1 : 0.0f;
      st[(mbase + r) * 32 + col]      = (_Float16)u0;
      st[(mbase + r) * 32 + col + 16] = (_Float16)u1;
    }
    __builtin_amdgcn_wave_barrier();
    asm volatile("s_wait_dscnt 0x0" ::: "memory");

    v8h lo = *(const v8h*)(st + col * 32 + kb);
    v8h hi = *(const v8h*)(st + col * 32 + kb + 8);
    v16h A2;
#pragma unroll
    for (int i = 0; i < 8; ++i) { A2[i] = lo[i]; A2[8 + i] = hi[i]; }
    int4 di0 = *(const int4*)(db + mbase);
    int4 di1 = *(const int4*)(db + mbase + 4);
    __builtin_amdgcn_wave_barrier();

    v8f d0 = wmma16(A2, Wb_lo, zero8());
    v8f d1 = wmma16(A2, Wb_hi, zero8());

    int dr[8] = {di0.x, di0.y, di0.z, di0.w, di1.x, di1.y, di1.z, di1.w};
#pragma unroll
    for (int r = 0; r < 8; ++r) {
      float u0 = d0[r] + bias2_lo;
      float u1 = d1[r] + bias2_hi;
      unsigned* row = (unsigned*)(h2 + (long)dr[r] * 32);
      if (u0 > 0.0f) atomicMax(row + col,      __float_as_uint(u0));
      if (u1 > 0.0f) atomicMax(row + col + 16, __float_as_uint(u1));
    }
  }
}

// Per-graph max pool (inputs >= 0, g init 0 => exact seg_max0 semantics)
__global__ void pool_kernel(const float* __restrict__ h2,
                            const int* __restrict__ batch,
                            float* __restrict__ g, int n) {
  long i = (long)blockIdx.x * blockDim.x + threadIdx.x;
  long stride = (long)gridDim.x * blockDim.x;
  long total = (long)n * 32;
  for (; i < total; i += stride) {
    int node = (int)(i >> 5), c = (int)(i & 31);
    float v = h2[i];
    if (v > 0.0f)
      atomicMax((unsigned*)&g[(long)batch[node] * 32 + c], __float_as_uint(v));
  }
}

// Head: out[64,3] = g[64,32] @ wc[32,3] + bc
__global__ void head_kernel(const float* __restrict__ g,
                            const float* __restrict__ wc,
                            const float* __restrict__ bc,
                            float* __restrict__ out) {
  int i = threadIdx.x;
  if (i < 64 * 3) {
    int gi = i / 3, j = i % 3;
    float acc = bc[j];
#pragma unroll
    for (int k = 0; k < 32; ++k) acc += g[gi * 32 + k] * wc[k * 3 + j];
    out[i] = acc;
  }
}

extern "C" void kernel_launch(void* const* d_in, const int* in_sizes, int n_in,
                              void* d_out, int out_size, void* d_ws, size_t ws_size,
                              hipStream_t stream) {
  const float* pos = (const float*)d_in[0];
  const float* w1a = (const float*)d_in[1];
  const float* b1a = (const float*)d_in[2];
  const float* w1b = (const float*)d_in[3];
  const float* b1b = (const float*)d_in[4];
  const float* w2a = (const float*)d_in[5];
  const float* b2a = (const float*)d_in[6];
  const float* w2b = (const float*)d_in[7];
  const float* b2b = (const float*)d_in[8];
  const float* wc  = (const float*)d_in[9];
  const float* bc  = (const float*)d_in[10];
  const int* ei    = (const int*)d_in[11];
  const int* batch = (const int*)d_in[12];

  const int N = in_sizes[0] / 2;       // nodes (pos is N x 2)
  const int E = in_sizes[11] / 2;      // edges (edge_index is 2 x E)
  const int G = 64;

  // workspace: h1[N*32], h2[N*32], g[G*32]  (~25.6 MB)
  float* h1 = (float*)d_ws;
  float* h2 = h1 + (size_t)N * 32;
  float* g  = h2 + (size_t)N * 32;

  zero_kernel<<<512, 256, 0, stream>>>(h1, (long)N * 64 + (long)G * 32);

  const int ntiles = (E + 15) / 16;
  layer1_kernel<<<1024, 256, 0, stream>>>(pos, w1a, b1a, w1b, b1b,
                                          ei, ei + E, h1, ntiles, E);
  layer2_kernel<<<1024, 256, 0, stream>>>(pos, h1, w2a, b2a, w2b, b2b,
                                          ei, ei + E, h2, ntiles, E);
  pool_kernel<<<512, 256, 0, stream>>>(h2, batch, g, N);
  head_kernel<<<1, 192, 0, stream>>>(g, wc, bc, (float*)d_out);
}
